// AttentionRope_23295902614141
// MI455X (gfx1250) — compile-verified, matrix operand order checked by reading, not run on hardware
//
#include <hip/hip_runtime.h>
#include <hip/hip_bf16.h>
#include <math.h>

typedef __bf16 bf16_t;
typedef __attribute__((ext_vector_type(8)))  __bf16 v8bf;
typedef __attribute__((ext_vector_type(16))) __bf16 v16bf;
typedef __attribute__((ext_vector_type(8)))  float  v8f;

#define NTOK 4096          // B*N
#define CDIM 1024
#define QKVDIM 3072
#define NHEAD 16
#define DH 64
#define NSEQ 2048
#define KTILES (CDIM / 32)

#if defined(__has_builtin)
#if __has_builtin(__builtin_amdgcn_tensor_load_to_lds)
#define USE_TDM 1
#endif
#endif
#ifndef USE_TDM
#define USE_TDM 0
#endif

static __device__ __forceinline__ v16bf join8(const bf16_t* p0, const bf16_t* p1) {
  v8bf lo = *(const v8bf*)p0;
  v8bf hi = *(const v8bf*)p1;
  return __builtin_shufflevector(lo, hi, 0,1,2,3,4,5,6,7,8,9,10,11,12,13,14,15);
}

static __device__ __forceinline__ v8f wmma_bf16(v16bf a, v16bf b, v8f c) {
  return __builtin_amdgcn_wmma_f32_16x16x32_bf16(false, a, false, b, (short)0, c, false, false);
}

#if USE_TDM
typedef __attribute__((ext_vector_type(4))) unsigned int u32x4;
typedef __attribute__((ext_vector_type(4))) int i32x4;
typedef __attribute__((ext_vector_type(8))) int i32x8;

// 2D bf16 tile (tile_w x tile_h, row-major, row stride in elements) -> LDS byte offset.
static __device__ __forceinline__ void tdm_load_tile(unsigned lds_byte_off,
                                                     const bf16_t* gptr,
                                                     unsigned tile_w, unsigned tile_h,
                                                     unsigned stride_elems) {
  unsigned long long ga = (unsigned long long)(uintptr_t)gptr;
  u32x4 g0;
  g0.x = 1u;                                              // count=1, user descriptor
  g0.y = lds_byte_off;                                    // lds_addr
  g0.z = (unsigned)ga;                                    // global_addr[31:0]
  g0.w = (unsigned)((ga >> 32) & 0x01FFFFFFull) | (2u << 30);  // addr hi | type=2
  i32x8 g1;
  g1[0] = 1 << 16;                                        // data_size = 2 bytes
  g1[1] = (int)((stride_elems & 0xFFFFu) << 16);          // tensor_dim0[15:0]
  g1[2] = (int)((stride_elems >> 16) | (0xFFFFu << 16));  // dim0 hi | tensor_dim1 lo
  g1[3] = (int)(tile_w << 16);                            // tensor_dim1 hi=0 | tile_dim0
  g1[4] = (int)(tile_h & 0xFFFFu);                        // tile_dim1 (tile_dim2=0)
  g1[5] = (int)stride_elems;                              // tensor_dim0_stride[31:0]
  g1[6] = 0;
  g1[7] = 0;
  i32x4 z4 = {0, 0, 0, 0};
#if __clang_major__ >= 23
  i32x8 z8 = {0, 0, 0, 0, 0, 0, 0, 0};
  __builtin_amdgcn_tensor_load_to_lds(g0, g1, z4, z4, z8, 0);
#else
  __builtin_amdgcn_tensor_load_to_lds(g0, g1, z4, z4, 0);
#endif
}
#endif

// ---------------- fp32 -> bf16 conversion, 8 elements per thread ----------------
__global__ __launch_bounds__(256) void cvt_bf16_v8(const float* __restrict__ s,
                                                   bf16_t* __restrict__ d, int n8) {
  int i = blockIdx.x * 256 + threadIdx.x;
  if (i < n8) {
    const float4* sp = (const float4*)s + (size_t)i * 2;
    float4 a = sp[0], b = sp[1];
    v8bf o;
    o[0] = (bf16_t)a.x; o[1] = (bf16_t)a.y; o[2] = (bf16_t)a.z; o[3] = (bf16_t)a.w;
    o[4] = (bf16_t)b.x; o[5] = (bf16_t)b.y; o[6] = (bf16_t)b.z; o[7] = (bf16_t)b.w;
    *((v8bf*)d + i) = o;
  }
}

// ---------------- tiled GEMM: out = A @ W^T (A:[M,1024], W:[N,1024], bf16) --------
// Block tile 256(M) x 128(N), 8 waves (4M x 2N), each wave 64x64 = 4x4 frags
// -> 16 WMMAs per 16 ds_load_b128 per K-step (DS-bandwidth balanced).
// TDM double-buffered staging: wave 0 prefetches tile kk+1 while all compute kk.
// QKV mode: bf16 out (ldo=3072), cols<1024 scaled by 0.125*log2(e) (exp2 softmax).
template <bool QKV>
__global__ __launch_bounds__(256) void gemm_wmma(const bf16_t* __restrict__ A,
                                                 const bf16_t* __restrict__ W,
                                                 bf16_t* __restrict__ outb,
                                                 float* __restrict__ outf,
                                                 const float* __restrict__ bias) {
  // per buffer: A tile 256x32 (8192 el) then B tile 128x32 (4096 el)
  __shared__ __align__(16) bf16_t smem[2][12288];

  const int tid  = threadIdx.x;
  const int lane = tid & 31;
  const int wave = tid >> 5;
  const int wm   = wave & 3;          // M offset wm*64
  const int wn   = wave >> 2;         // N offset wn*64
  const int ln   = lane & 15;
  const int half = lane >> 4;
  const int mb   = blockIdx.y;
  const int nb   = blockIdx.x;
  const int ldo  = QKV ? QKVDIM : CDIM;
  const size_t abase = (size_t)(mb * 256) * CDIM;
  const size_t bbase = (size_t)(nb * 128) * CDIM;

  v8f acc[4][4];
#pragma unroll
  for (int i = 0; i < 4; ++i)
#pragma unroll
    for (int j = 0; j < 4; ++j)
#pragma unroll
      for (int r = 0; r < 8; ++r) acc[i][j][r] = 0.0f;

#if USE_TDM
  if (wave == 0) {
    tdm_load_tile(0u, A + abase, 32, 256, CDIM);
    tdm_load_tile(16384u, W + bbase, 32, 128, CDIM);
    __builtin_amdgcn_s_wait_tensorcnt(0);
  }
  __syncthreads();
#endif

  for (int kk = 0; kk < KTILES; ++kk) {
    const int cur = kk & 1;
#if USE_TDM
    if (wave == 0 && kk + 1 < KTILES) {  // prefetch next tile, overlaps WMMAs below
      const unsigned nxt = (unsigned)(cur ^ 1) * 24576u;
      tdm_load_tile(nxt, A + abase + (kk + 1) * 32, 32, 256, CDIM);
      tdm_load_tile(nxt + 16384u, W + bbase + (kk + 1) * 32, 32, 128, CDIM);
    }
    const bf16_t* As = &smem[cur][0];
    const bf16_t* Bs = &smem[cur][8192];
#else
    const int k0 = kk * 32;
#pragma unroll
    for (int u = tid; u < 1536; u += 256) {
      if (u < 1024) {
        int row = u >> 2, seg = u & 3;
        *(v8bf*)&smem[0][row * 32 + seg * 8] =
            *(const v8bf*)&A[abase + (size_t)row * CDIM + k0 + seg * 8];
      } else {
        int ub = u - 1024, row = ub >> 2, seg = ub & 3;
        *(v8bf*)&smem[0][8192 + row * 32 + seg * 8] =
            *(const v8bf*)&W[bbase + (size_t)row * CDIM + k0 + seg * 8];
      }
    }
    __syncthreads();
    const bf16_t* As = &smem[0][0];
    const bf16_t* Bs = &smem[0][8192];
    (void)cur;
#endif

    v16bf af[4], bf[4];
#pragma unroll
    for (int i = 0; i < 4; ++i) {
      int base = (wm * 64 + i * 16 + ln) * 32 + half * 8;
      af[i] = join8(&As[base], &As[base + 16]);
    }
#pragma unroll
    for (int j = 0; j < 4; ++j) {
      int base = (wn * 64 + j * 16 + ln) * 32 + half * 8;
      bf[j] = join8(&Bs[base], &Bs[base + 16]);
    }
#pragma unroll
    for (int i = 0; i < 4; ++i)
#pragma unroll
      for (int j = 0; j < 4; ++j) acc[i][j] = wmma_bf16(af[i], bf[j], acc[i][j]);

#if USE_TDM
    if (wave == 0) __builtin_amdgcn_s_wait_tensorcnt(0);
#endif
    __syncthreads();
  }

#pragma unroll
  for (int i = 0; i < 4; ++i) {
#pragma unroll
    for (int j = 0; j < 4; ++j) {
      int gcol  = nb * 128 + wn * 64 + j * 16 + ln;
      int grow0 = mb * 256 + wm * 64 + i * 16 + half * 8;
#pragma unroll
      for (int r = 0; r < 8; ++r) {
        float v = acc[i][j][r];
        if (QKV) {
          if (gcol < CDIM) v *= 0.125f * 1.4426950408889634f;  // Dh^-.5 * log2(e)
          outb[(size_t)(grow0 + r) * ldo + gcol] = (bf16_t)v;
        } else {
          outf[(size_t)(grow0 + r) * ldo + gcol] = v + bias[gcol];
        }
      }
    }
  }
}

// ---------------- fused flash attention (exp2-based, KV block = 64) ----------------
// grid: (N/64 q-blocks, B*H). 128 threads = 4 waves, each wave owns 16 q rows.
// Row sums computed as P x ones via WMMA (no sum butterflies).
__global__ __launch_bounds__(128) void flash_attn(const bf16_t* __restrict__ qkv,
                                                  bf16_t* __restrict__ aout) {
  __shared__ __align__(16) bf16_t Qs[64 * 64];
  __shared__ __align__(16) bf16_t Ks[64 * 64];       // [j][d]
  __shared__ __align__(16) bf16_t VTs[64 * 64];      // [d][j]
  __shared__ __align__(16) bf16_t Ps[4][16 * 64];    // per-wave P tile

  const int tid  = threadIdx.x;
  const int lane = tid & 31;
  const int wave = tid >> 5;
  const int ln   = lane & 15;
  const int half = lane >> 4;
  const int qb   = blockIdx.x;
  const int bh   = blockIdx.y;
  const int h    = bh & (NHEAD - 1);
  const int b    = bh >> 4;
  const int rb   = b * NSEQ;
  const int qcol = h * DH;
  const int kcol = CDIM + h * DH;
  const int vcol = 2 * CDIM + h * DH;

#pragma unroll
  for (int u = tid; u < 512; u += 128) {
    int row = u >> 3, seg = u & 7;
    *(v8bf*)&Qs[row * 64 + seg * 8] =
        *(const v8bf*)&qkv[(size_t)(rb + qb * 64 + row) * QKVDIM + qcol + seg * 8];
  }
  __syncthreads();

  v16bf qa[2];
#pragma unroll
  for (int t = 0; t < 2; ++t) {
    int base = (wave * 16 + ln) * 64 + t * 32 + half * 8;
    qa[t] = join8(&Qs[base], &Qs[base + 16]);
  }

  v16bf ones;
#pragma unroll
  for (int e = 0; e < 16; ++e) ones[e] = (bf16_t)1.0f;

  v8f o[4];
  float rm[8], rs[8], sc[8];
#pragma unroll
  for (int t = 0; t < 4; ++t)
#pragma unroll
    for (int r = 0; r < 8; ++r) o[t][r] = 0.0f;
#pragma unroll
  for (int r = 0; r < 8; ++r) { rm[r] = -1e30f; rs[r] = 0.0f; }

  for (int jb = 0; jb < NSEQ / 64; ++jb) {
    __syncthreads();
    // stage K [64 j x 64 d] row-major; V transposed -> VTs [64 d x 64 j]
#pragma unroll
    for (int u = tid; u < 512; u += 128) {
      int row = u >> 3, seg = u & 7;
      *(v8bf*)&Ks[row * 64 + seg * 8] =
          *(const v8bf*)&qkv[(size_t)(rb + jb * 64 + row) * QKVDIM + kcol + seg * 8];
      v8bf vv =
          *(const v8bf*)&qkv[(size_t)(rb + jb * 64 + row) * QKVDIM + vcol + seg * 8];
#pragma unroll
      for (int e = 0; e < 8; ++e) VTs[(seg * 8 + e) * 64 + row] = vv[e];
    }
    __syncthreads();

    // S = Q K^T : four 16x16 tiles (cols jb*64 + c*16)
    v8f s[4];
#pragma unroll
    for (int c = 0; c < 4; ++c) {
#pragma unroll
      for (int r = 0; r < 8; ++r) s[c][r] = 0.0f;
#pragma unroll
      for (int t = 0; t < 2; ++t) {
        int base = (c * 16 + ln) * 64 + t * 32 + half * 8;
        s[c] = wmma_bf16(qa[t], join8(&Ks[base], &Ks[base + 16]), s[c]);
      }
    }

    // online softmax: row = half*8 + r lives on the 16 lanes of this half
#pragma unroll
    for (int r = 0; r < 8; ++r) {
      float bm = fmaxf(fmaxf(s[0][r], s[1][r]), fmaxf(s[2][r], s[3][r]));
#pragma unroll
      for (int msk = 1; msk < 16; msk <<= 1) bm = fmaxf(bm, __shfl_xor(bm, msk, 32));
      float nm = fmaxf(rm[r], bm);
      sc[r] = exp2f(rm[r] - nm);  // rm init -1e30 -> underflows to 0
      rm[r] = nm;
#pragma unroll
      for (int c = 0; c < 4; ++c) {
        float p = exp2f(s[c][r] - nm);
        Ps[wave][(half * 8 + r) * 64 + c * 16 + ln] = (bf16_t)p;
      }
#pragma unroll
      for (int t = 0; t < 4; ++t) o[t][r] *= sc[r];
    }
    __syncthreads();

    v16bf pa[2];
#pragma unroll
    for (int kc = 0; kc < 2; ++kc) {
      int base = ln * 64 + kc * 32 + half * 8;
      pa[kc] = join8(&Ps[wave][base], &Ps[wave][base + 16]);
    }
    // row sums via WMMA: bs = P x ones, every lane's element r = rowsum(half*8+r)
    v8f bs;
#pragma unroll
    for (int r = 0; r < 8; ++r) bs[r] = 0.0f;
    bs = wmma_bf16(pa[0], ones, bs);
    bs = wmma_bf16(pa[1], ones, bs);
#pragma unroll
    for (int r = 0; r < 8; ++r) rs[r] = rs[r] * sc[r] + bs[r];

#pragma unroll
    for (int t = 0; t < 4; ++t)
#pragma unroll
      for (int kc = 0; kc < 2; ++kc) {
        int vb = (t * 16 + ln) * 64 + kc * 32 + half * 8;
        o[t] = wmma_bf16(pa[kc], join8(&VTs[vb], &VTs[vb + 16]), o[t]);
      }
  }

#pragma unroll
  for (int r = 0; r < 8; ++r) sc[r] = __builtin_amdgcn_rcpf(rs[r]);
#pragma unroll
  for (int t = 0; t < 4; ++t) {
    int gcol = h * DH + t * 16 + ln;
#pragma unroll
    for (int r = 0; r < 8; ++r) {
      int grow = b * NSEQ + qb * 64 + wave * 16 + half * 8 + r;
      aout[(size_t)grow * CDIM + gcol] = (bf16_t)(o[t][r] * sc[r]);
    }
  }
}

// ---------------- launch ----------------
extern "C" void kernel_launch(void* const* d_in, const int* in_sizes, int n_in,
                              void* d_out, int out_size, void* d_ws, size_t ws_size,
                              hipStream_t stream) {
  const float* x      = (const float*)d_in[0];  // [2,2048,1024]
  const float* w_qkv  = (const float*)d_in[1];  // [3072,1024]
  const float* w_proj = (const float*)d_in[2];  // [1024,1024]
  const float* b_proj = (const float*)d_in[3];  // [1024]
  float* out = (float*)d_out;

  char* ws = (char*)d_ws;
  bf16_t* xb    = (bf16_t*)(ws);                               // 8 MB
  bf16_t* wqkvb = (bf16_t*)(ws + (size_t)8 * 1024 * 1024);     // 6 MB
  bf16_t* wprjb = (bf16_t*)(ws + (size_t)14 * 1024 * 1024);    // 2 MB
  bf16_t* qkvb  = (bf16_t*)(ws + (size_t)16 * 1024 * 1024);    // 24 MB
  bf16_t* aoutb = (bf16_t*)(ws + (size_t)40 * 1024 * 1024);    // 8 MB

  const int nx = NTOK * CDIM, nwq = QKVDIM * CDIM, nwp = CDIM * CDIM;
  cvt_bf16_v8<<<nx / 8 / 256, 256, 0, stream>>>(x, xb, nx / 8);
  cvt_bf16_v8<<<nwq / 8 / 256, 256, 0, stream>>>(w_qkv, wqkvb, nwq / 8);
  cvt_bf16_v8<<<nwp / 8 / 256, 256, 0, stream>>>(w_proj, wprjb, nwp / 8);

  gemm_wmma<true><<<dim3(QKVDIM / 128, NTOK / 256), 256, 0, stream>>>(
      xb, wqkvb, qkvb, nullptr, nullptr);

  flash_attn<<<dim3(NSEQ / 64, 2 * NHEAD), 128, 0, stream>>>(qkvb, aoutb);

  gemm_wmma<false><<<dim3(CDIM / 128, NTOK / 256), 256, 0, stream>>>(
      aoutb, wprjb, nullptr, out, b_proj);
}
